// PAC_17394617548977
// MI455X (gfx1250) — matrix operand order repaired and checked by reading, not compiled, not run on hardware
//
#include <hip/hip_runtime.h>

// ---------------- problem constants ----------------
#define T_FULL   2048
#define T_TRIM   1536
#define NTAPS    257
#define NF       80
#define NPHA     50
#define NAMP     30
#define NBINS    18
#define NSIG     256                 // B*C*G = 2*16*8
#define KSTEPS1  9                   // ceil(257/32) -> K padded to 288
#define KSTEPS2  48                  // 1536/32
#define NT2      96                  // 1536/16 time tiles
#define NROWS2   (NSIG*NF)           // 20480 (signal,filter) rows
#define MT2      (NROWS2/16)         // 1280 row tiles (80 = 5*16 -> no signal straddle)
#define KBP_H    (5*KSTEPS1*32*16)   // packed FIR fragment halves (23040 = 45KB)
#define PI_F     3.14159265358979323846f

typedef __attribute__((ext_vector_type(16))) __bf16 v16bf;
typedef __attribute__((ext_vector_type(8)))  float  v8f;

union BFrag { v16bf v; unsigned short u[16]; uint4 q[2]; };

static __device__ __forceinline__ unsigned short f2bf(float f) {
  union { float f; unsigned int u; } c; c.f = f;
  unsigned int u = c.u + 0x7FFFu + ((c.u >> 16) & 1u);   // RNE
  return (unsigned short)(u >> 16);
}
static __device__ __forceinline__ float bf2f(unsigned short b) {
  union { unsigned int u; float f; } c; c.u = ((unsigned int)b) << 16;
  return c.f;
}

// ---------------- prep kernels ----------------
__global__ void k_prep_xbf(const float* __restrict__ x,
                           unsigned short* __restrict__ xbf, int n) {
  int i = blockIdx.x * blockDim.x + threadIdx.x;
  if (i < n) xbf[i] = f2bf(x[i]);
}

// FIR kernels packed directly into the WMMA 16-bit B (32x16) lane layout:
// lane l: column n = l%16 (filter), k = 8*(l/16) + h + 8*(h/8)  (h = half index 0..15)
__global__ void k_prep_kpack(const float* __restrict__ kern,
                             unsigned short* __restrict__ kbp) {
  int i = blockIdx.x * blockDim.x + threadIdx.x;
  if (i >= KBP_H) return;
  int h    = i & 15;
  int lane = (i >> 4) & 31;
  int ks   = (i >> 9) % KSTEPS1;
  int ft   = (i >> 9) / KSTEPS1;
  int f    = ft * 16 + (lane & 15);
  int tap  = ks * 32 + 8 * (lane >> 4) + h + 8 * (h >> 3);
  float v  = (tap < NTAPS) ? kern[f * NTAPS + tap] : 0.f;
  kbp[i] = f2bf(v);
}

// Circular discrete-Hilbert circulant H[tau][t] = hhat[(t - tau) mod N],
// hhat[n] = (2/N) cot(pi n / N) for odd n, 0 otherwise  (exact FFT-mask operator)
// packed into the WMMA B (32x16) lane layout.
__global__ void k_prep_hpack(unsigned short* __restrict__ hpk) {
  int i = blockIdx.x * blockDim.x + threadIdx.x;
  if (i >= NT2 * KSTEPS2 * 32 * 16) return;
  int h    = i & 15;
  int lane = (i >> 4) & 31;
  int ks   = (i >> 9) % KSTEPS2;
  int nt   = (i >> 9) / KSTEPS2;
  int t    = nt * 16 + (lane & 15);
  int tau  = ks * 32 + 8 * (lane >> 4) + h + 8 * (h >> 3);
  int d    = t - tau; d %= T_TRIM; if (d < 0) d += T_TRIM;
  float v = 0.f;
  if (d & 1) {
    double a = 3.14159265358979323846 * (double)d / (double)T_TRIM;
    v = (float)((2.0 / (double)T_TRIM) * (cos(a) / sin(a)));
  }
  hpk[i] = f2bf(v);
}

// ---------------- stage 1: bandpass GEMM (WMMA bf16, all-LDS operands) -----
// D(16t x 16f) += A(16t x 32k) * B(32k x 16f). Signal row AND all packed FIR
// fragments staged in LDS (49 KB) -> inner loop is ds_load_b128 + v_wmma only.
__global__ void __launch_bounds__(256)
k_bandpass(const unsigned short* __restrict__ xbf,
           const unsigned short* __restrict__ kbp,
           unsigned short* __restrict__ fil) {
  __shared__ unsigned short xsh[T_FULL];   // 4 KB signal row
  __shared__ unsigned short ksh[KBP_H];    // 45 KB packed FIR fragments
  int tid  = threadIdx.x;
  int lane = tid & 31;
  int wv   = tid >> 5;
  int s    = blockIdx.x;

  { // stage signal row + all kernel fragments
    const uint2* xr = (const uint2*)(xbf + (size_t)s * T_FULL);
    uint2* xs = (uint2*)xsh;
    for (int i = tid; i < T_FULL / 4; i += 256) xs[i] = xr[i];
    const uint4* kr = (const uint4*)kbp;
    uint4* kd = (uint4*)ksh;
    for (int i = tid; i < KBP_H / 8; i += 256) kd[i] = kr[i];
  }
  __syncthreads();

  int tt   = blockIdx.y * 8 + wv;   // 0..95 time tile
  int t0   = tt * 16;
  int m    = lane & 15;
  int half = lane >> 4;

  v8f acc[5];
  #pragma unroll
  for (int ft = 0; ft < 5; ++ft)
    #pragma unroll
    for (int r = 0; r < 8; ++r) acc[ft][r] = 0.f;

  #pragma unroll
  for (int ks = 0; ks < KSTEPS1; ++ks) {
    // A (16-bit 16x32) lane layout: row m = l%16, k runs [8*half,+8) and [16+8*half,+8)
    BFrag a;
    int base = t0 + 128 + m + ks * 32 + 8 * half;
    #pragma unroll
    for (int j = 0; j < 8; ++j) a.u[j]     = xsh[base + j];
    #pragma unroll
    for (int j = 0; j < 8; ++j) a.u[8 + j] = xsh[base + 16 + j];
    BFrag b[5];
    #pragma unroll
    for (int ft = 0; ft < 5; ++ft) {
      const uint4* bp = (const uint4*)(ksh + (((ft * KSTEPS1) + ks) * 32 + lane) * 16);
      b[ft].q[0] = bp[0]; b[ft].q[1] = bp[1];
    }
    #pragma unroll
    for (int ft = 0; ft < 5; ++ft)
      acc[ft] = __builtin_amdgcn_wmma_f32_16x16x32_bf16(
          false, a.v, false, b[ft].v, (short)0, acc[ft], false, false);
  }
  // D layout: VGPR r -> time t0 + r + 8*half, filter ft*16 + (lane&15)
  #pragma unroll
  for (int ft = 0; ft < 5; ++ft) {
    int f = ft * 16 + m;
    unsigned short* dst = fil + ((size_t)(s * NF + f)) * T_TRIM + t0 + 8 * half;
    #pragma unroll
    for (int r = 0; r < 8; ++r) dst[r] = f2bf(acc[ft][r]);
  }
}

// ---------------- stage 2: Hilbert circulant GEMM (double-buffered) --------
static __device__ __forceinline__ void s2_load(BFrag& a, BFrag b[4],
    const unsigned short* __restrict__ yr, const unsigned short* __restrict__ hw,
    int ks, int half, int ng) {
  const unsigned short* ap = yr + ks * 32 + 8 * half;
  a.q[0] = *(const uint4*)(ap);
  a.q[1] = *(const uint4*)(ap + 16);
  #pragma unroll
  for (int q = 0; q < 4; ++q) {
    const uint4* bp = (const uint4*)(hw + (size_t)((ng * 4 + q) * KSTEPS2 + ks) * 512);
    b[q].q[0] = bp[0]; b[q].q[1] = bp[1];
  }
}
static __device__ __forceinline__ void s2_mma(v8f acc[4], const BFrag& a, const BFrag b[4]) {
  #pragma unroll
  for (int q = 0; q < 4; ++q)
    acc[q] = __builtin_amdgcn_wmma_f32_16x16x32_bf16(
        false, a.v, false, b[q].v, (short)0, acc[q], false, false);
}

__global__ void __launch_bounds__(256)
k_hilbert(const unsigned short* __restrict__ fil,
          const unsigned short* __restrict__ hpk,
          unsigned char* __restrict__ pidx,
          float* __restrict__ aamp) {
  int tid  = threadIdx.x;
  int lane = tid & 31;
  int wv   = tid >> 5;
  int mt   = blockIdx.x;               // 0..1279 row tile
  int ng   = blockIdx.y * 8 + wv;      // 0..23 group of 4 time tiles
  int m    = lane & 15;
  int half = lane >> 4;

  int row0 = mt * 16;
  const unsigned short* yr = fil + (size_t)(row0 + m) * T_TRIM;
  const unsigned short* hw = hpk + (size_t)lane * 16;

  v8f acc[4];
  #pragma unroll
  for (int q = 0; q < 4; ++q)
    #pragma unroll
    for (int r = 0; r < 8; ++r) acc[q][r] = 0.f;

  // 2-deep software pipeline: loads for kstep k+1 issue before WMMAs of kstep k
  BFrag a0, a1, b0[4], b1[4];
  s2_load(a0, b0, yr, hw, 0, half, ng);
  for (int ks = 0; ks < KSTEPS2; ks += 2) {
    s2_load(a1, b1, yr, hw, ks + 1, half, ng);
    s2_mma(acc, a0, b0);
    if (ks + 2 < KSTEPS2) s2_load(a0, b0, yr, hw, ks + 2, half, ng);
    s2_mma(acc, a1, b1);
  }

  // epilogue: row tiles never straddle a signal (80 = 5*16)
  int sig   = row0 / NF;
  int fbase = row0 - sig * NF;
  #pragma unroll
  for (int q = 0; q < 4; ++q) {
    int t = (ng * 4 + q) * 16 + m;
    #pragma unroll
    for (int r = 0; r < 8; ++r) {
      int rr  = r + 8 * half;
      int f   = fbase + rr;
      int row = row0 + rr;
      float im = acc[q][r];
      float re = bf2f(fil[(size_t)row * T_TRIM + t]);   // Re(analytic) == y
      if (f < NPHA) {
        float ph = atan2f(im, re);
        int b = (int)floorf((ph + PI_F) * ((float)NBINS / (2.0f * PI_F)));
        b = b < 0 ? 0 : (b > NBINS - 1 ? NBINS - 1 : b);
        pidx[((size_t)sig * NPHA + f) * T_TRIM + t] = (unsigned char)b;
      } else {
        aamp[((size_t)sig * NAMP + (f - NPHA)) * T_TRIM + t] = sqrtf(re * re + im * im);
      }
    }
  }
}

// ---------------- stage 3: modulation index ----------------
__global__ void __launch_bounds__(256)
k_mi_hist(const unsigned char* __restrict__ pidx,
          const float* __restrict__ aamp,
          float* __restrict__ mi5) {
  __shared__ float sums[NBINS];
  __shared__ unsigned int cnts[NBINS];
  __shared__ unsigned char ib[T_TRIM];
  int tid = threadIdx.x;
  int s = blockIdx.x, p = blockIdx.y;
  const unsigned char* ir = pidx + ((size_t)s * NPHA + p) * T_TRIM;
  for (int i = tid; i < T_TRIM; i += 256) ib[i] = ir[i];
  if (tid < NBINS) cnts[tid] = 0u;
  __syncthreads();
  for (int i = tid; i < T_TRIM; i += 256) atomicAdd(&cnts[ib[i]], 1u);
  __syncthreads();
  for (int a = 0; a < NAMP; ++a) {
    if (tid < NBINS) sums[tid] = 0.f;
    __syncthreads();
    const float* ar = aamp + ((size_t)s * NAMP + a) * T_TRIM;
    for (int i = tid; i < T_TRIM; i += 256) atomicAdd(&sums[ib[i]], ar[i]);
    __syncthreads();
    if (tid == 0) {
      float mean[NBINS]; float tot = 0.f;
      #pragma unroll
      for (int b = 0; b < NBINS; ++b) { mean[b] = sums[b] / ((float)cnts[b] + 1e-9f); tot += mean[b]; }
      float inv = 1.0f / (tot + 1e-9f);
      float hs = 0.f;
      #pragma unroll
      for (int b = 0; b < NBINS; ++b) { float pr = mean[b] * inv; hs += pr * logf(pr + 1e-9f); }
      float ln_n = logf((float)NBINS);
      mi5[((size_t)s * NPHA + p) * NAMP + a] = (ln_n + hs) / ln_n;
    }
    __syncthreads();
  }
}

__global__ void k_mi_reduce(const float* __restrict__ mi5, float* __restrict__ out) {
  int i = blockIdx.x * blockDim.x + threadIdx.x;
  if (i >= 32 * NPHA * NAMP) return;
  int bc  = i / (NPHA * NAMP);
  int rem = i - bc * (NPHA * NAMP);
  int p = rem / NAMP, a = rem - (rem / NAMP) * NAMP;
  float acc = 0.f;
  #pragma unroll
  for (int g = 0; g < 8; ++g)
    acc += mi5[(((size_t)(bc * 8 + g) * NPHA) + p) * NAMP + a];
  out[i] = acc * 0.125f;
}

// ---------------- launcher ----------------
extern "C" void kernel_launch(void* const* d_in, const int* in_sizes, int n_in,
                              void* d_out, int out_size, void* d_ws, size_t ws_size,
                              hipStream_t stream) {
  (void)in_sizes; (void)n_in; (void)out_size; (void)ws_size;
  const float* x    = (const float*)d_in[0];
  const float* kern = (const float*)d_in[1];
  float* out        = (float*)d_out;

  char* ws = (char*)d_ws;
  size_t off = 0;
  auto take = [&](size_t bytes) {
    char* p = ws + off;
    off += bytes;
    off = (off + 255) & ~(size_t)255;
    return p;
  };
  unsigned short* xbf  = (unsigned short*)take((size_t)NSIG * T_FULL * 2);          // 1 MB
  unsigned short* kbp  = (unsigned short*)take((size_t)KBP_H * 2);                  // 45 KB
  unsigned short* hpk  = (unsigned short*)take((size_t)NT2 * KSTEPS2 * 32 * 16 * 2);// 4.5 MB
  unsigned short* fil  = (unsigned short*)take((size_t)NROWS2 * T_TRIM * 2);        // 60 MB
  unsigned char*  pidx = (unsigned char*) take((size_t)NSIG * NPHA * T_TRIM);       // 18.8 MB
  float*          aamp = (float*)         take((size_t)NSIG * NAMP * T_TRIM * 4);   // 45 MB
  float*          mi5  = (float*)         take((size_t)NSIG * NPHA * NAMP * 4);     // 1.5 MB

  k_prep_xbf  <<<(NSIG * T_FULL + 255) / 256, 256, 0, stream>>>(x, xbf, NSIG * T_FULL);
  k_prep_kpack<<<(KBP_H + 255) / 256, 256, 0, stream>>>(kern, kbp);
  k_prep_hpack<<<(NT2 * KSTEPS2 * 32 * 16 + 255) / 256, 256, 0, stream>>>(hpk);
  k_bandpass  <<<dim3(NSIG, 12), 256, 0, stream>>>(xbf, kbp, fil);
  k_hilbert   <<<dim3(MT2, 3), 256, 0, stream>>>(fil, hpk, pidx, aamp);
  k_mi_hist   <<<dim3(NSIG, NPHA), 256, 0, stream>>>(pidx, aamp, mi5);
  k_mi_reduce <<<(32 * NPHA * NAMP + 255) / 256, 256, 0, stream>>>(mi5, out);
}
